// SwinDecoderBlock_609885356486
// MI455X (gfx1250) — compile-verified, matrix operand order checked by reading, not run on hardware
//
#include <hip/hip_runtime.h>
#include <hip/hip_bf16.h>

typedef __attribute__((ext_vector_type(16))) _Float16 v16h;
typedef __attribute__((ext_vector_type(8)))  _Float16 v8h;
typedef __attribute__((ext_vector_type(8)))  float    v8f;
typedef __attribute__((ext_vector_type(4)))  float    v4f;

#define DEV __device__ __forceinline__

// ---------------- problem sizes ----------------
constexpr int  BATCH = 8;
constexpr int  DIM   = 256;
constexpr int  INDIM = 512;
constexpr int  HIN   = 48,  WIN_ = 48;
constexpr int  HOUT  = 96,  WOUT = 96;
constexpr int  HP    = 98,  WP   = 98;     // padded to multiple of 7
constexpr int  HIDDEN= 1024;
constexpr int  WSZ   = 7,   NWIN = 14, NTOK = 49;
constexpr int  HEADS = 8,   HD   = 32;
constexpr long P_    = (long)BATCH*HOUT*WOUT;   // 73728 real pixels
constexpr long PP_   = (long)BATCH*HP*WP;       // 76832 padded pixels
constexpr long PIN_  = (long)BATCH*HIN*WIN_;    // 18432 input pixels

// ---------------- workspace layout (bytes) ----------------
constexpr size_t ALN(size_t x){ return (x + 255) & ~(size_t)255; }
constexpr size_t SZ_XH   = ALN((size_t)PIN_*INDIM*2);   // x as rows, f16
constexpr size_t SZ_SKH  = ALN((size_t)P_*DIM*2);       // skip as rows, f16
constexpr size_t OFF_XH  = 0;
constexpr size_t OFF_SKH = OFF_XH + SZ_XH;
constexpr size_t OFF_OBUF= 0;                           // alias: consumed before obuf written
constexpr size_t OFF_WUP = OFF_SKH + SZ_SKH;            // 4 x (256x512) f16
constexpr size_t OFF_WSK = OFF_WUP + ALN((size_t)4*DIM*INDIM*2);
constexpr size_t OFF_WQKV= OFF_WSK + ALN((size_t)DIM*DIM*2);
constexpr size_t OFF_WPRJ= OFF_WQKV+ ALN((size_t)3*DIM*DIM*2);
constexpr size_t OFF_WM1 = OFF_WPRJ+ ALN((size_t)DIM*DIM*2);
constexpr size_t OFF_WM2 = OFF_WM1 + ALN((size_t)HIDDEN*DIM*2);
constexpr size_t OFF_BIAS= OFF_WM2 + ALN((size_t)DIM*HIDDEN*2);
constexpr size_t OFF_SC  = OFF_BIAS+ ALN((size_t)HEADS*NTOK*NTOK*4); // shortcut f32
constexpr size_t OFF_YS  = OFF_SC  + ALN((size_t)P_*DIM*4);          // ysum f32 (alias x2)
constexpr size_t OFF_X2  = OFF_YS;
constexpr size_t OFF_XNP = OFF_YS  + ALN((size_t)P_*DIM*4);          // padded LN1 out f16 (alias xn2)
constexpr size_t OFF_XN2 = OFF_XNP;
constexpr size_t OFF_QKV = OFF_XNP + ALN((size_t)PP_*DIM*2);         // qkv f16 (alias mlp hidden)
constexpr size_t OFF_HH  = OFF_QKV;

// ---------------- WMMA helpers ----------------
DEV v16h frag_ld(const _Float16* rowp, int koff) {
  union { v16h v; struct { v8h lo, hi; } s; } u;
  u.s.lo = *(const v8h*)(rowp + koff);
  u.s.hi = *(const v8h*)(rowp + koff + 16);
  return u.v;
}
DEV v8f wmma32(v16h a, v16h b, v8f c) {
  return __builtin_amdgcn_wmma_f32_16x16x32_f16(false, a, false, b, (short)0, c, false, false);
}
DEV float wave_red_sum(float v){
  #pragma unroll
  for (int m = 16; m >= 1; m >>= 1) v += __shfl_xor(v, m, 32);
  return v;
}
#define VZERO {0.f,0.f,0.f,0.f,0.f,0.f,0.f,0.f}

// ---------------- elementwise / staging kernels ----------------
__global__ __launch_bounds__(256) void k_nchw_to_rows(const float* __restrict__ src,
                                                      _Float16* __restrict__ dst,
                                                      int Cc, int HWp, long total) {
  long idx = (long)blockIdx.x * 256 + threadIdx.x;
  if (idx >= total) return;
  long b = idx / ((long)Cc * HWp);
  long r = idx % ((long)Cc * HWp);
  int  c = (int)(r / HWp);
  int  p = (int)(r % HWp);
  dst[((long)b * HWp + p) * Cc + c] = (_Float16)src[idx];
}

__global__ __launch_bounds__(256) void k_cvt_f16(const float* __restrict__ s,
                                                 _Float16* __restrict__ d, long n) {
  long idx = (long)blockIdx.x * 256 + threadIdx.x;
  if (idx < n) d[idx] = (_Float16)s[idx];
}

// up_w (512,256,2,2) -> 4 matrices [d][o*512+c]
__global__ __launch_bounds__(256) void k_pack_upw(const float* __restrict__ upw,
                                                  _Float16* __restrict__ w4) {
  long idx = (long)blockIdx.x * 256 + threadIdx.x;
  long total = 4L * DIM * INDIM;
  if (idx >= total) return;
  int d = (int)(idx / (DIM * INDIM));
  int rem = (int)(idx % (DIM * INDIM));
  int o = rem / INDIM, c = rem % INDIM;
  int di = d >> 1, dj = d & 1;
  w4[idx] = (_Float16)upw[((long)c * DIM + o) * 4 + di * 2 + dj];
}

__global__ __launch_bounds__(256) void k_zero16(uint4* __restrict__ p, long n16) {
  long idx = (long)blockIdx.x * 256 + threadIdx.x;
  if (idx < n16) p[idx] = make_uint4(0u, 0u, 0u, 0u);
}

// rel-pos bias gather: biasf[h][n][m]
__global__ __launch_bounds__(256) void k_bias(const float* __restrict__ rpb,
                                              float* __restrict__ biasf) {
  int idx = blockIdx.x * 256 + threadIdx.x;
  if (idx >= HEADS * NTOK * NTOK) return;
  int h = idx / (NTOK * NTOK);
  int rem = idx % (NTOK * NTOK);
  int n = rem / NTOK, m = rem % NTOK;
  int dr = n / WSZ - m / WSZ + (WSZ - 1);
  int dc = n % WSZ - m % WSZ + (WSZ - 1);
  biasf[idx] = rpb[(dr * (2 * WSZ - 1) + dc) * HEADS + h];
}

// LayerNorm over C=256; one wave per row; optional padded-grid output addressing
__global__ __launch_bounds__(128) void k_layernorm(const float* __restrict__ in,
                                                   const float* __restrict__ g,
                                                   const float* __restrict__ be,
                                                   _Float16* __restrict__ out,
                                                   int padded) {
  int lane = threadIdx.x & 31, wv = threadIdx.x >> 5;
  long row = (long)blockIdx.x * 4 + wv;
  const float* x = in + row * DIM + lane * 8;
  float v[8];
  v4f a0 = *(const v4f*)(x);
  v4f a1 = *(const v4f*)(x + 4);
  #pragma unroll
  for (int i = 0; i < 4; ++i) { v[i] = a0[i]; v[i + 4] = a1[i]; }
  float s = 0.f, s2 = 0.f;
  #pragma unroll
  for (int i = 0; i < 8; ++i) { s += v[i]; s2 += v[i] * v[i]; }
  s = wave_red_sum(s); s2 = wave_red_sum(s2);
  float mean = s * (1.f / DIM);
  float var  = s2 * (1.f / DIM) - mean * mean;
  float rs   = rsqrtf(var + 1e-6f);
  long orow;
  if (padded) {
    int b = (int)(row / (HOUT * WOUT));
    int rem = (int)(row % (HOUT * WOUT));
    int y = rem / WOUT, xx = rem % WOUT;
    orow = ((long)(b * HP + y) * WP + xx);
  } else {
    orow = row;
  }
  _Float16* o = out + orow * DIM + lane * 8;
  #pragma unroll
  for (int i = 0; i < 8; ++i) {
    int c = lane * 8 + i;
    o[i] = (_Float16)((v[i] - mean) * rs * g[c] + be[c]);
  }
}

// ---------------- GEMM kernels: 32x32 tile per wave (2x2 WMMA register blocking) ----------------
// inner K-step macro: 8 b128 loads -> 4 independent WMMAs
#define GEMM_LOOP(ap0, ap1, bp0, bp1, KDIM)                         \
  v8f c00 = VZERO, c01 = VZERO, c10 = VZERO, c11 = VZERO;           \
  for (int k0 = 0; k0 < (KDIM); k0 += 32) {                         \
    int koff = k0 + hf * 8;                                         \
    v16h a0 = frag_ld(ap0, koff);                                   \
    v16h a1 = frag_ld(ap1, koff);                                   \
    v16h b0 = frag_ld(bp0, koff);                                   \
    v16h b1 = frag_ld(bp1, koff);                                   \
    c00 = wmma32(a0, b0, c00);                                      \
    c01 = wmma32(a0, b1, c01);                                      \
    c10 = wmma32(a1, b0, c10);                                      \
    c11 = wmma32(a1, b1, c11);                                      \
  }

// ConvT k2s2: for each (di,dj) sub-matrix, C = xh(18432x512) * Wd^T + up_b, scattered to 96x96 grid
__global__ __launch_bounds__(128) void k_upconv(const _Float16* __restrict__ xh,
                                                const _Float16* __restrict__ w4,
                                                const float* __restrict__ upb,
                                                float* __restrict__ shortcut) {
  int lane = threadIdx.x & 31, wv = threadIdx.x >> 5;
  long wid = (long)blockIdx.x * 4 + wv;          // 576*8*4 waves
  int nt = (int)(wid % 8);
  long t = wid / 8;
  int mt = (int)(t % 576);
  int d  = (int)(t / 576);
  int l = lane & 15, hf = lane >> 4;
  const _Float16* ap0 = xh + (long)(mt * 32 + l) * INDIM;
  const _Float16* ap1 = ap0 + 16L * INDIM;
  const _Float16* bp0 = w4 + (long)d * DIM * INDIM + (long)(nt * 32 + l) * INDIM;
  const _Float16* bp1 = bp0 + 16L * INDIM;
  GEMM_LOOP(ap0, ap1, bp0, bp1, INDIM)
  int di = d >> 1, dj = d & 1;
  v8f cc[2][2] = {{c00, c01}, {c10, c11}};
  #pragma unroll
  for (int ti = 0; ti < 2; ++ti)
    #pragma unroll
    for (int tj = 0; tj < 2; ++tj) {
      int col = nt * 32 + tj * 16 + l;
      float bias = upb[col];
      #pragma unroll
      for (int r = 0; r < 8; ++r) {
        int rl = mt * 32 + ti * 16 + r + hf * 8;
        int b = rl / (HIN * WIN_), rem = rl % (HIN * WIN_);
        int i = rem / WIN_, j = rem % WIN_;
        long orow = (long)b * (HOUT * WOUT) + (long)(2 * i + di) * WOUT + (2 * j + dj);
        shortcut[orow * DIM + col] = cc[ti][tj][r] + bias;
      }
    }
}

// ysum = shortcut + skip_rows * skip_w^T
__global__ __launch_bounds__(128) void k_skip(const _Float16* __restrict__ skh,
                                              const _Float16* __restrict__ wsk,
                                              const float* __restrict__ shortcut,
                                              float* __restrict__ ysum) {
  int lane = threadIdx.x & 31, wv = threadIdx.x >> 5;
  long wid = (long)blockIdx.x * 4 + wv;          // 2304*8 waves
  int nt = (int)(wid % 8);
  int mt = (int)(wid / 8);
  int l = lane & 15, hf = lane >> 4;
  const _Float16* ap0 = skh + (long)(mt * 32 + l) * DIM;
  const _Float16* ap1 = ap0 + 16L * DIM;
  const _Float16* bp0 = wsk + (long)(nt * 32 + l) * DIM;
  const _Float16* bp1 = bp0 + 16L * DIM;
  GEMM_LOOP(ap0, ap1, bp0, bp1, DIM)
  v8f cc[2][2] = {{c00, c01}, {c10, c11}};
  #pragma unroll
  for (int ti = 0; ti < 2; ++ti)
    #pragma unroll
    for (int tj = 0; tj < 2; ++tj) {
      int col = nt * 32 + tj * 16 + l;
      #pragma unroll
      for (int r = 0; r < 8; ++r) {
        long row = (long)(mt * 32 + ti * 16 + r + hf * 8);
        ysum[row * DIM + col] = shortcut[row * DIM + col] + cc[ti][tj][r];
      }
    }
}

// qkv = xnp(padded,76832x256) * qkv_w^T + qkv_b  -> f16 (PPx768)
__global__ __launch_bounds__(128) void k_qkv(const _Float16* __restrict__ xnp,
                                             const _Float16* __restrict__ wq,
                                             const float* __restrict__ qb,
                                             _Float16* __restrict__ qkv) {
  int lane = threadIdx.x & 31, wv = threadIdx.x >> 5;
  long wid = (long)blockIdx.x * 4 + wv;          // 2401*24 waves
  int nt = (int)(wid % 24);
  int mt = (int)(wid / 24);
  int l = lane & 15, hf = lane >> 4;
  const _Float16* ap0 = xnp + (long)(mt * 32 + l) * DIM;
  const _Float16* ap1 = ap0 + 16L * DIM;
  const _Float16* bp0 = wq  + (long)(nt * 32 + l) * DIM;
  const _Float16* bp1 = bp0 + 16L * DIM;
  GEMM_LOOP(ap0, ap1, bp0, bp1, DIM)
  v8f cc[2][2] = {{c00, c01}, {c10, c11}};
  #pragma unroll
  for (int ti = 0; ti < 2; ++ti)
    #pragma unroll
    for (int tj = 0; tj < 2; ++tj) {
      int col = nt * 32 + tj * 16 + l;
      float bias = qb[col];
      #pragma unroll
      for (int r = 0; r < 8; ++r) {
        long row = (long)(mt * 32 + ti * 16 + r + hf * 8);
        qkv[row * (3 * DIM) + col] = (_Float16)(cc[ti][tj][r] + bias);
      }
    }
}

// windowed attention: one wave per (window, head, 16-query tile)
__global__ __launch_bounds__(128) void k_attn(const _Float16* __restrict__ qkv,
                                              const float* __restrict__ biasf,
                                              _Float16* __restrict__ obuf) {
  __shared__ __attribute__((aligned(16))) _Float16 lds_p [4][16][64];
  __shared__ __attribute__((aligned(16))) _Float16 lds_vT[4][32][72];
  int lane = threadIdx.x & 31, wv = threadIdx.x >> 5;
  long wid = (long)blockIdx.x * 4 + wv;          // 1568*8*4 waves
  int qt  = (int)(wid & 3);
  int h   = (int)((wid >> 2) & 7);
  int win = (int)(wid >> 5);
  int b = win / (NWIN * NWIN);
  int wrem = win % (NWIN * NWIN);
  int wi = wrem / NWIN, wj = wrem % NWIN;
  int l = lane & 15, hf = lane >> 4;
  const float scale = 0.17677669529663687f;

  // ---- Q fragment ----
  int qi = qt * 16 + l; int qic = qi < NTOK ? qi : NTOK - 1;
  long qpix = ((long)(b * HP + wi * WSZ + qic / WSZ)) * WP + (wj * WSZ + qic % WSZ);
  const _Float16* qrow = qkv + qpix * (3 * DIM) + h * HD;
  v16h aq = frag_ld(qrow, hf * 8);

  // ---- S = Q K^T (4 column tiles of 16) ----
  v8f s[4];
  #pragma unroll
  for (int nt = 0; nt < 4; ++nt) {
    int kj = nt * 16 + l; int kjc = kj < NTOK ? kj : NTOK - 1;
    long kpix = ((long)(b * HP + wi * WSZ + kjc / WSZ)) * WP + (wj * WSZ + kjc % WSZ);
    const _Float16* krow = qkv + kpix * (3 * DIM) + DIM + h * HD;
    v16h bk = frag_ld(krow, hf * 8);
    v8f z = VZERO;
    s[nt] = wmma32(aq, bk, z);
  }
  // scale + rel-pos bias + key-pad mask
  #pragma unroll
  for (int nt = 0; nt < 4; ++nt) {
    #pragma unroll
    for (int r = 0; r < 8; ++r) {
      int qr = qt * 16 + r + 8 * hf;
      int kj = nt * 16 + l;
      float val = s[nt][r] * scale;
      if (kj < NTOK && qr < NTOK) val += biasf[(h * NTOK + qr) * NTOK + kj];
      if (kj >= NTOK) val = -1e30f;
      s[nt][r] = val;
    }
  }
  // ---- softmax per row (cols across 16 lanes of each half + 4 tiles) ----
  #pragma unroll
  for (int r = 0; r < 8; ++r) {
    float m = s[0][r];
    #pragma unroll
    for (int nt = 1; nt < 4; ++nt) m = fmaxf(m, s[nt][r]);
    #pragma unroll
    for (int off = 8; off >= 1; off >>= 1) m = fmaxf(m, __shfl_xor(m, off, 32));
    float sum = 0.f;
    #pragma unroll
    for (int nt = 0; nt < 4; ++nt) { float e = __expf(s[nt][r] - m); s[nt][r] = e; sum += e; }
    #pragma unroll
    for (int off = 8; off >= 1; off >>= 1) sum += __shfl_xor(sum, off, 32);
    float inv = 1.f / sum;
    #pragma unroll
    for (int nt = 0; nt < 4; ++nt) s[nt][r] *= inv;
  }
  // ---- stage P (f16) and V^T into LDS ----
  #pragma unroll
  for (int nt = 0; nt < 4; ++nt)
    #pragma unroll
    for (int r = 0; r < 8; ++r)
      lds_p[wv][r + 8 * hf][nt * 16 + l] = (_Float16)s[nt][r];
  #pragma unroll
  for (int t = 0; t < 2; ++t) {
    int key = lane * 2 + t; int keyc = key < NTOK ? key : NTOK - 1;
    long vpix = ((long)(b * HP + wi * WSZ + keyc / WSZ)) * WP + (wj * WSZ + keyc % WSZ);
    const _Float16* vrow = qkv + vpix * (3 * DIM) + 2 * DIM + h * HD;
    v8h a0 = *(const v8h*)(vrow);
    v8h a1 = *(const v8h*)(vrow + 8);
    v8h a2 = *(const v8h*)(vrow + 16);
    v8h a3 = *(const v8h*)(vrow + 24);
    #pragma unroll
    for (int d = 0; d < 8; ++d) {
      lds_vT[wv][d     ][key] = a0[d];
      lds_vT[wv][d +  8][key] = a1[d];
      lds_vT[wv][d + 16][key] = a2[d];
      lds_vT[wv][d + 24][key] = a3[d];
    }
  }
  __syncthreads();
  // ---- O = P V (K=64 in two steps, N=32 in two tiles) ----
  v8f o0 = VZERO;
  v8f o1 = VZERO;
  #pragma unroll
  for (int ks = 0; ks < 2; ++ks) {
    int koff = ks * 32 + hf * 8;
    v16h apv = frag_ld(&lds_p[wv][l][0], koff);
    v16h bv0 = frag_ld(&lds_vT[wv][l     ][0], koff);
    v16h bv1 = frag_ld(&lds_vT[wv][l + 16][0], koff);
    o0 = wmma32(apv, bv0, o0);
    o1 = wmma32(apv, bv1, o1);
  }
  // ---- store O (f16, padded-grid layout, heads concatenated) ----
  #pragma unroll
  for (int r = 0; r < 8; ++r) {
    int qr = qt * 16 + r + 8 * hf;
    if (qr < NTOK) {
      long opix = ((long)(b * HP + wi * WSZ + qr / WSZ)) * WP + (wj * WSZ + qr % WSZ);
      obuf[opix * DIM + h * HD + l]      = (_Float16)o0[r];
      obuf[opix * DIM + h * HD + 16 + l] = (_Float16)o1[r];
    }
  }
}

// x2 = shortcut + obuf * proj_w^T + proj_b
__global__ __launch_bounds__(128) void k_proj(const _Float16* __restrict__ obuf,
                                              const _Float16* __restrict__ wp,
                                              const float* __restrict__ pb,
                                              const float* __restrict__ shortcut,
                                              float* __restrict__ x2) {
  int lane = threadIdx.x & 31, wv = threadIdx.x >> 5;
  long wid = (long)blockIdx.x * 4 + wv;          // 2304*8 waves
  int nt = (int)(wid % 8);
  int mt = (int)(wid / 8);
  int l = lane & 15, hf = lane >> 4;
  // A rows: real pixel -> padded-grid address (two row bases)
  int rl0 = mt * 32 + l;
  int rl1 = rl0 + 16;
  int b0_ = rl0 / (HOUT * WOUT), re0 = rl0 % (HOUT * WOUT);
  int b1_ = rl1 / (HOUT * WOUT), re1 = rl1 % (HOUT * WOUT);
  const _Float16* ap0 = obuf + ((long)(b0_ * HP + re0 / WOUT) * WP + re0 % WOUT) * DIM;
  const _Float16* ap1 = obuf + ((long)(b1_ * HP + re1 / WOUT) * WP + re1 % WOUT) * DIM;
  const _Float16* bp0 = wp + (long)(nt * 32 + l) * DIM;
  const _Float16* bp1 = bp0 + 16L * DIM;
  GEMM_LOOP(ap0, ap1, bp0, bp1, DIM)
  v8f cc[2][2] = {{c00, c01}, {c10, c11}};
  #pragma unroll
  for (int ti = 0; ti < 2; ++ti)
    #pragma unroll
    for (int tj = 0; tj < 2; ++tj) {
      int col = nt * 32 + tj * 16 + l;
      float bias = pb[col];
      #pragma unroll
      for (int r = 0; r < 8; ++r) {
        long row = (long)(mt * 32 + ti * 16 + r + 8 * hf);
        x2[row * DIM + col] = shortcut[row * DIM + col] + cc[ti][tj][r] + bias;
      }
    }
}

// h = gelu(xn2 * w1^T + b1)  -> f16 (P x 1024)
__global__ __launch_bounds__(128) void k_fc1(const _Float16* __restrict__ xn2,
                                             const _Float16* __restrict__ w1,
                                             const float* __restrict__ b1,
                                             _Float16* __restrict__ hh) {
  int lane = threadIdx.x & 31, wv = threadIdx.x >> 5;
  long wid = (long)blockIdx.x * 4 + wv;          // 2304*32 waves
  int nt = (int)(wid % 32);
  int mt = (int)(wid / 32);
  int l = lane & 15, hf = lane >> 4;
  const _Float16* ap0 = xn2 + (long)(mt * 32 + l) * DIM;
  const _Float16* ap1 = ap0 + 16L * DIM;
  const _Float16* bp0 = w1  + (long)(nt * 32 + l) * DIM;
  const _Float16* bp1 = bp0 + 16L * DIM;
  GEMM_LOOP(ap0, ap1, bp0, bp1, DIM)
  v8f cc[2][2] = {{c00, c01}, {c10, c11}};
  #pragma unroll
  for (int ti = 0; ti < 2; ++ti)
    #pragma unroll
    for (int tj = 0; tj < 2; ++tj) {
      int col = nt * 32 + tj * 16 + l;
      float bias = b1[col];
      #pragma unroll
      for (int r = 0; r < 8; ++r) {
        long row = (long)(mt * 32 + ti * 16 + r + 8 * hf);
        float v = cc[ti][tj][r] + bias;
        float g = 0.5f * v * (1.f + erff(v * 0.70710678118654752f)); // exact GELU
        hh[row * HIDDEN + col] = (_Float16)g;
      }
    }
}

// out(NCHW) = x2 + h * w2^T + b2
__global__ __launch_bounds__(128) void k_fc2(const _Float16* __restrict__ hh,
                                             const _Float16* __restrict__ w2,
                                             const float* __restrict__ b2,
                                             const float* __restrict__ x2,
                                             float* __restrict__ out) {
  int lane = threadIdx.x & 31, wv = threadIdx.x >> 5;
  long wid = (long)blockIdx.x * 4 + wv;          // 2304*8 waves
  int nt = (int)(wid % 8);
  int mt = (int)(wid / 8);
  int l = lane & 15, hf = lane >> 4;
  const _Float16* ap0 = hh + (long)(mt * 32 + l) * HIDDEN;
  const _Float16* ap1 = ap0 + 16L * HIDDEN;
  const _Float16* bp0 = w2 + (long)(nt * 32 + l) * HIDDEN;
  const _Float16* bp1 = bp0 + 16L * HIDDEN;
  GEMM_LOOP(ap0, ap1, bp0, bp1, HIDDEN)
  v8f cc[2][2] = {{c00, c01}, {c10, c11}};
  #pragma unroll
  for (int ti = 0; ti < 2; ++ti)
    #pragma unroll
    for (int tj = 0; tj < 2; ++tj) {
      int col = nt * 32 + tj * 16 + l;
      float bias = b2[col];
      #pragma unroll
      for (int r = 0; r < 8; ++r) {
        long row = (long)(mt * 32 + ti * 16 + r + 8 * hf);
        int bb = (int)(row / (HOUT * WOUT));
        int rem = (int)(row % (HOUT * WOUT));
        int y = rem / WOUT, xx = rem % WOUT;
        float v = x2[row * DIM + col] + cc[ti][tj][r] + bias;
        out[(((long)bb * DIM + col) * HOUT + y) * WOUT + xx] = v;
      }
    }
}

// ---------------- launch ----------------
extern "C" void kernel_launch(void* const* d_in, const int* in_sizes, int n_in,
                              void* d_out, int out_size, void* d_ws, size_t ws_size,
                              hipStream_t stream) {
  const float* x      = (const float*)d_in[0];
  const float* skipf  = (const float*)d_in[1];
  const float* up_w   = (const float*)d_in[2];
  const float* up_b   = (const float*)d_in[3];
  const float* skip_w = (const float*)d_in[4];
  const float* gamma1 = (const float*)d_in[5];
  const float* beta1  = (const float*)d_in[6];
  const float* qkv_w  = (const float*)d_in[7];
  const float* qkv_b  = (const float*)d_in[8];
  const float* rpb    = (const float*)d_in[9];
  const float* proj_w = (const float*)d_in[10];
  const float* proj_b = (const float*)d_in[11];
  const float* gamma2 = (const float*)d_in[12];
  const float* beta2  = (const float*)d_in[13];
  const float* mlp_w1 = (const float*)d_in[14];
  const float* mlp_b1 = (const float*)d_in[15];
  const float* mlp_w2 = (const float*)d_in[16];
  const float* mlp_b2 = (const float*)d_in[17];

  char* ws = (char*)d_ws;
  _Float16* xh   = (_Float16*)(ws + OFF_XH);
  _Float16* skh  = (_Float16*)(ws + OFF_SKH);
  _Float16* obuf = (_Float16*)(ws + OFF_OBUF);
  _Float16* wup4 = (_Float16*)(ws + OFF_WUP);
  _Float16* wsk  = (_Float16*)(ws + OFF_WSK);
  _Float16* wqkv = (_Float16*)(ws + OFF_WQKV);
  _Float16* wprj = (_Float16*)(ws + OFF_WPRJ);
  _Float16* wm1  = (_Float16*)(ws + OFF_WM1);
  _Float16* wm2  = (_Float16*)(ws + OFF_WM2);
  float*    biasf= (float*)   (ws + OFF_BIAS);
  float*    shortcut = (float*)(ws + OFF_SC);
  float*    ysum = (float*)   (ws + OFF_YS);
  float*    x2   = (float*)   (ws + OFF_X2);
  _Float16* xnp  = (_Float16*)(ws + OFF_XNP);
  _Float16* xn2  = (_Float16*)(ws + OFF_XN2);
  _Float16* qkvb = (_Float16*)(ws + OFF_QKV);
  _Float16* hh   = (_Float16*)(ws + OFF_HH);

  auto blk = [](long n, int t){ return dim3((unsigned)((n + t - 1) / t)); };

  // stage activations / weights as f16
  k_nchw_to_rows<<<blk(PIN_*INDIM,256),256,0,stream>>>(x, xh, INDIM, HIN*WIN_, PIN_*INDIM);
  k_nchw_to_rows<<<blk(P_*DIM,256),256,0,stream>>>(skipf, skh, DIM, HOUT*WOUT, P_*DIM);
  k_pack_upw<<<blk(4L*DIM*INDIM,256),256,0,stream>>>(up_w, wup4);
  k_cvt_f16<<<blk((long)DIM*DIM,256),256,0,stream>>>(skip_w, wsk, (long)DIM*DIM);
  k_cvt_f16<<<blk(3L*DIM*DIM,256),256,0,stream>>>(qkv_w, wqkv, 3L*DIM*DIM);
  k_cvt_f16<<<blk((long)DIM*DIM,256),256,0,stream>>>(proj_w, wprj, (long)DIM*DIM);
  k_cvt_f16<<<blk((long)HIDDEN*DIM,256),256,0,stream>>>(mlp_w1, wm1, (long)HIDDEN*DIM);
  k_cvt_f16<<<blk((long)DIM*HIDDEN,256),256,0,stream>>>(mlp_w2, wm2, (long)DIM*HIDDEN);
  k_bias<<<blk(HEADS*NTOK*NTOK,256),256,0,stream>>>(rpb, biasf);

  // upsample conv-transpose (4 sub-GEMMs, 32x32 tiles) -> shortcut
  k_upconv<<<dim3(576*8*4/4),128,0,stream>>>(xh, wup4, up_b, shortcut);
  // skip adapter 1x1 -> ysum
  k_skip<<<dim3(2304*8/4),128,0,stream>>>(skh, wsk, shortcut, ysum);
  // LN1 into zero-padded 98x98 grid
  k_zero16<<<blk((long)PP_*DIM*2/16,256),256,0,stream>>>((uint4*)xnp, (long)PP_*DIM*2/16);
  k_layernorm<<<dim3((unsigned)(P_/4)),128,0,stream>>>(ysum, gamma1, beta1, xnp, 1);
  // QKV projection on padded grid
  k_qkv<<<dim3((unsigned)(2401*24/4)),128,0,stream>>>(xnp, wqkv, qkv_b, qkvb);
  // windowed attention
  k_attn<<<dim3(1568*8*4/4),128,0,stream>>>(qkvb, biasf, obuf);
  // output projection + residual with shortcut
  k_proj<<<dim3(2304*8/4),128,0,stream>>>(obuf, wprj, proj_b, shortcut, x2);
  // LN2 + MLP + residual, final NCHW store
  k_layernorm<<<dim3((unsigned)(P_/4)),128,0,stream>>>(x2, gamma2, beta2, xn2, 0);
  k_fc1<<<dim3(2304*32/4),128,0,stream>>>(xn2, wm1, mlp_b1, hh);
  k_fc2<<<dim3(2304*8/4),128,0,stream>>>(hh, wm2, mlp_b2, x2, (float*)d_out);
}
